// Encoder_55233279426649
// MI455X (gfx1250) — matrix operand order repaired and checked by reading, not compiled, not run on hardware
//
#include <hip/hip_runtime.h>

// ---------------------------------------------------------------------------
// MI455X (gfx1250) fused encoder: bf16 WMMA pipeline (wave32).
//   K1: qkv = x@Wqkv+b (and l@Wqkv1+b1)  -> q,k head-major bf16, v transposed
//   K2: per-(b,h,16 rows): dots/dots1 WMMA -> fused conv-BN-leaky-conv 'sup'
//       -> masked softmax (f32 in LDS, probs rewritten bf16 in place)
//       -> attn@v / attn1@v1 WMMA -> head outputs bf16
//   K3: out = O@Wmlp+b (and O1@Wmlp1+b1) -> fp32 into d_out (concatenated)
// ---------------------------------------------------------------------------

typedef __bf16 bf16;
typedef __attribute__((ext_vector_type(16))) __bf16 v16bf;
typedef __attribute__((ext_vector_type(8)))  __bf16 v8bf;
typedef __attribute__((ext_vector_type(8)))  float  v8f;

#define DEVFN __device__ __forceinline__

#define B_    32
#define N_    257
#define D_    512
#define H_    8
#define DH_   64
#define NP_   288                       // N padded to multiple of 32
#define M_    (B_ * N_)                 // 8224 rows
#define SCALE_ 0.125f                   // (D/H)^-0.5 = 64^-0.5

static const size_t SETQ = (size_t)B_ * H_ * NP_ * DH_;  // per-stream q/k/vT elems
static const size_t SETO = (size_t)M_ * D_;              // per-stream head-out elems

// ---------------------------------------------------------------------------
// WMMA helper:  D = A(16x32 bf16) x B(32x16 bf16) + C(f32)
DEVFN v8f wmma_bf16(v16bf a, v16bf b, v8f c) {
    return __builtin_amdgcn_wmma_f32_16x16x32_bf16(
        /*neg_a=*/false, a, /*neg_b=*/false, b,
        /*c_mod=*/(short)0, c, /*reuse_a=*/false, /*reuse_b=*/false);
}

// A-operand fragment from bf16 row-major row: elements {kb8..kb8+7, 16+kb8..+7}
DEVFN v16bf frag_A_bf16(const bf16* p, int kb8) {
    v8bf lo = *(const v8bf*)(p + kb8);
    v8bf hi = *(const v8bf*)(p + 16 + kb8);
    return __builtin_shufflevector(lo, hi, 0,1,2,3,4,5,6,7,8,9,10,11,12,13,14,15);
}

// A-operand fragment from fp32 row (convert to bf16 on the fly)
DEVFN v16bf frag_A_f32(const float* p, int kb8) {
    const float4 f0 = *(const float4*)(p + kb8);
    const float4 f1 = *(const float4*)(p + kb8 + 4);
    const float4 f2 = *(const float4*)(p + 16 + kb8);
    const float4 f3 = *(const float4*)(p + 16 + kb8 + 4);
    v16bf a;
    a[0]=(bf16)f0.x;  a[1]=(bf16)f0.y;  a[2]=(bf16)f0.z;  a[3]=(bf16)f0.w;
    a[4]=(bf16)f1.x;  a[5]=(bf16)f1.y;  a[6]=(bf16)f1.z;  a[7]=(bf16)f1.w;
    a[8]=(bf16)f2.x;  a[9]=(bf16)f2.y;  a[10]=(bf16)f2.z; a[11]=(bf16)f2.w;
    a[12]=(bf16)f3.x; a[13]=(bf16)f3.y; a[14]=(bf16)f3.z; a[15]=(bf16)f3.w;
    return a;
}

// ---------------------------------------------------------------------------
// Kernel 1: QKV projection GEMM.  grid = (24 ntiles, 129 mtiles, 2 streams)
// block = 128 (4 waves); each wave: 16 rows x 64 cols, K loop in chunks of 32.
__global__ __launch_bounds__(128)
void qkv_gemm_kernel(const float* __restrict__ x, const float* __restrict__ l,
                     const float* __restrict__ Wqkv,  const float* __restrict__ bqkv,
                     const float* __restrict__ Wqkv1, const float* __restrict__ bqkv1,
                     bf16* __restrict__ wsQ, bf16* __restrict__ wsK,
                     bf16* __restrict__ wsVt) {
    __shared__ __align__(32) bf16 Bst[4 * 32 * 16];   // 4 col-subtiles in B-frag layout

    const int tid  = threadIdx.x;
    const int w    = tid >> 5, lane = tid & 31;
    const int r16  = lane & 15;
    const int kb8  = (lane >> 4) << 3;
    const int set  = blockIdx.z;
    const int nbase = blockIdx.x * 64;
    const int mbase = blockIdx.y * 64 + (w << 4);

    const float* A    = set ? l      : x;
    const float* W    = set ? Wqkv1  : Wqkv;
    const float* bias = set ? bqkv1  : bqkv;

    v8f acc[4];
    for (int t = 0; t < 4; ++t)
        for (int i = 0; i < 8; ++i) acc[t][i] = 0.f;

    const int mA = mbase + r16;
    const float* arow = A + (size_t)(mA < M_ ? mA : 0) * D_;

    for (int kb = 0; kb < D_; kb += 32) {
        // Stage W[kb..kb+31][nbase..nbase+63] in B-fragment layout (bf16)
        for (int idx = tid; idx < 2048; idx += 128) {
            int kk = idx >> 6, nn = idx & 63;
            float wv = W[(size_t)(kb + kk) * (3 * D_) + nbase + nn];
            Bst[(((nn >> 4) << 5) + (nn & 15) + ((kk & 8) << 1)) * 16 + ((kk & 7) | ((kk >> 1) & 8))] = (bf16)wv;
        }
        __syncthreads();
        v16bf a = frag_A_f32(arow + kb, kb8);
#pragma unroll
        for (int t = 0; t < 4; ++t) {
            v16bf b = *(const v16bf*)(Bst + ((t << 5) + lane) * 16);
            acc[t] = wmma_bf16(a, b, acc[t]);
        }
        __syncthreads();
    }

    // Epilogue: bias add, scatter to q/k (head-major, Npad) and v (transposed)
    const size_t setoff = set ? SETQ : 0;
#pragma unroll
    for (int t = 0; t < 4; ++t) {
        const int c    = nbase + (t << 4) + r16;      // column in [0,1536)
        const int part = c >> 9;                      // 0=q 1=k 2=v
        const int rem  = c & 511;
        const int hh   = rem >> 6, dd = rem & 63;
        const float bv = bias[c];
#pragma unroll
        for (int r = 0; r < 8; ++r) {
            const int mo = mbase + ((lane >> 4) << 3) + r;
            if (mo < M_) {
                const int bb = mo / N_;
                const int n  = mo - bb * N_;
                const bf16 val = (bf16)(acc[t][r] + bv);
                if (part == 0)
                    wsQ[setoff + ((size_t)(bb * H_ + hh) * NP_ + n) * DH_ + dd] = val;
                else if (part == 1)
                    wsK[setoff + ((size_t)(bb * H_ + hh) * NP_ + n) * DH_ + dd] = val;
                else
                    wsVt[setoff + ((size_t)(bb * H_ + hh) * DH_ + dd) * NP_ + n] = val;
            }
        }
    }
}

// ---------------------------------------------------------------------------
// Kernel 2: fused attention.  grid = (17 row-tiles, B*H)  block = 128 (4 waves)
__global__ __launch_bounds__(128)
void attn_kernel(const bf16* __restrict__ wsQ, const bf16* __restrict__ wsK,
                 const bf16* __restrict__ wsVt, bf16* __restrict__ wsO,
                 const float* __restrict__ c1w, const float* __restrict__ c1b,
                 const float* __restrict__ bng, const float* __restrict__ bnb,
                 const float* __restrict__ c2w, const float* __restrict__ c2b) {
    __shared__ __align__(64) float S[2][16 * NP_];    // 36 KB score rows

    const int tid  = threadIdx.x;
    const int w    = tid >> 5, lane = tid & 31;
    const int r16  = lane & 15;
    const int kb8  = (lane >> 4) << 3;
    const int kb16 = (lane >> 4) << 4;
    const int bh   = blockIdx.y;
    const int b    = bh >> 3, h = bh & 7;
    const int nbase = blockIdx.x * 16;

    // Fold conv1 + BN(eval) into affine coeffs: y_o = A0*s + A1*s1 + Cc
    float A0[8], A1[8], Cc[8], w2[8];
    const float inv = rsqrtf(1.0f + 1e-5f);
#pragma unroll
    for (int o = 0; o < 8; ++o) {
        const float g = bng[h * 8 + o] * inv;
        A0[o] = c1w[(h * 8 + o) * 2 + 0] * g;
        A1[o] = c1w[(h * 8 + o) * 2 + 1] * g;
        Cc[o] = c1b[h * 8 + o] * g + bnb[h * 8 + o];
        w2[o] = c2w[h * 8 + o];
    }
    const float b2 = c2b[h];

    // Q fragments for this 16-row tile (both streams), K = 64 -> 2 chunks
    const size_t headQ = (size_t)(b * H_ + h) * NP_ * DH_;
    const bf16* q0r = wsQ + headQ + (size_t)(nbase + r16) * DH_;
    const bf16* q1r = q0r + SETQ;
    v16bf qa0[2], qa1[2];
#pragma unroll
    for (int c = 0; c < 2; ++c) {
        qa0[c] = frag_A_bf16(q0r + c * 32, kb8);
        qa1[c] = frag_A_bf16(q1r + c * 32, kb8);
    }

    // Scores + fused sup, masked, into LDS
    for (int jt = w; jt < 18; jt += 4) {
        const int jb = jt << 4;
        const bf16* k0r = wsK + headQ + (size_t)(jb + r16) * DH_;
        const bf16* k1r = k0r + SETQ;
        v8f d0, d1;
        for (int i = 0; i < 8; ++i) { d0[i] = 0.f; d1[i] = 0.f; }
#pragma unroll
        for (int c = 0; c < 2; ++c) {
            v16bf kf0 = frag_A_bf16(k0r + c * 32, kb8);
            v16bf kf1 = frag_A_bf16(k1r + c * 32, kb8);
            d0 = wmma_bf16(qa0[c], kf0, d0);
            d1 = wmma_bf16(qa1[c], kf1, d1);
        }
        const int  jj  = jb + r16;
        const bool vld = jj < N_;
#pragma unroll
        for (int r = 0; r < 8; ++r) {
            const int il = ((lane >> 4) << 3) + r;
            const float s  = d0[r] * SCALE_;
            const float s1 = d1[r] * SCALE_;
            float sup = b2;
#pragma unroll
            for (int o = 0; o < 8; ++o) {
                float y = fmaf(A0[o], s, fmaf(A1[o], s1, Cc[o]));
                y = (y > 0.f) ? y : 0.01f * y;
                sup = fmaf(y, w2[o], sup);
            }
            S[0][il * NP_ + jj] = vld ? (s  + sup) : -1e30f;
            S[1][il * NP_ + jj] = vld ? (s1 + sup) : -1e30f;
        }
    }
    __syncthreads();

    // Row softmax; rewrite probabilities as bf16 in place over the f32 rows
    if (tid < 32) {
        const int sel = tid >> 4, row = tid & 15;
        float* sr = &S[sel][row * NP_];
        float mx = -3.0e38f;
        for (int j = 0; j < NP_; ++j) mx = fmaxf(mx, sr[j]);
        float sum = 0.f;
        for (int j = 0; j < NP_; ++j) sum += __expf(sr[j] - mx);
        const float rs = 1.f / sum;
        bf16* pr = (bf16*)sr;   // byte 2j written after byte 4j read: safe
        for (int j = 0; j < NP_; ++j) pr[j] = (bf16)(__expf(sr[j] - mx) * rs);
    }
    __syncthreads();

    // attn @ v (each wave owns a 16-wide d-tile), K = 288 -> 9 chunks of 32
    const int dbase = w << 4;
    const size_t headV = (size_t)(b * H_ + h) * DH_ * NP_;
    const bf16* v0r = wsVt + headV + (size_t)(dbase + r16) * NP_;
    const bf16* v1r = v0r + SETQ;
    const bf16* p0  = (const bf16*)&S[0][r16 * NP_];
    const bf16* p1  = (const bf16*)&S[1][r16 * NP_];
    v8f o0, o1;
    for (int i = 0; i < 8; ++i) { o0[i] = 0.f; o1[i] = 0.f; }
#pragma unroll
    for (int kc = 0; kc < 9; ++kc) {
        const int jb = kc << 5;
        v16bf pa0 = frag_A_bf16(p0 + jb, kb8);
        v16bf pa1 = frag_A_bf16(p1 + jb, kb8);
        v16bf vf0 = frag_A_bf16(v0r + jb, kb8);
        v16bf vf1 = frag_A_bf16(v1r + jb, kb8);
        o0 = wmma_bf16(pa0, vf0, o0);
        o1 = wmma_bf16(pa1, vf1, o1);
    }
#pragma unroll
    for (int r = 0; r < 8; ++r) {
        const int il = ((lane >> 4) << 3) + r;
        const int n  = nbase + il;
        if (n < N_) {
            const int c = h * DH_ + dbase + r16;
            const size_t off = ((size_t)(b * N_ + n)) * D_ + c;
            wsO[off]        = (bf16)o0[r];
            wsO[SETO + off] = (bf16)o1[r];
        }
    }
}

// ---------------------------------------------------------------------------
// Kernel 3: output MLP GEMM.  grid = (8 ntiles, 129 mtiles, 2 streams)
__global__ __launch_bounds__(128)
void mlp_gemm_kernel(const bf16* __restrict__ wsO,
                     const float* __restrict__ Wm,  const float* __restrict__ bm,
                     const float* __restrict__ Wm1, const float* __restrict__ bm1,
                     float* __restrict__ out) {
    __shared__ __align__(32) bf16 Bst[4 * 32 * 16];

    const int tid  = threadIdx.x;
    const int w    = tid >> 5, lane = tid & 31;
    const int r16  = lane & 15;
    const int kb8  = (lane >> 4) << 3;
    const int set  = blockIdx.z;
    const int nbase = blockIdx.x * 64;
    const int mbase = blockIdx.y * 64 + (w << 4);

    const bf16*  A    = wsO + (set ? SETO : 0);
    const float* W    = set ? Wm1 : Wm;
    const float* bias = set ? bm1 : bm;
    float*       dst  = out + (set ? SETO : 0);

    v8f acc[4];
    for (int t = 0; t < 4; ++t)
        for (int i = 0; i < 8; ++i) acc[t][i] = 0.f;

    const int mA = mbase + r16;
    const bf16* arow = A + (size_t)(mA < M_ ? mA : 0) * D_;

    for (int kb = 0; kb < D_; kb += 32) {
        for (int idx = tid; idx < 2048; idx += 128) {
            int kk = idx >> 6, nn = idx & 63;
            float wv = W[(size_t)(kb + kk) * D_ + nbase + nn];
            Bst[(((nn >> 4) << 5) + (nn & 15) + ((kk & 8) << 1)) * 16 + ((kk & 7) | ((kk >> 1) & 8))] = (bf16)wv;
        }
        __syncthreads();
        v16bf a = frag_A_bf16(arow + kb, kb8);
#pragma unroll
        for (int t = 0; t < 4; ++t) {
            v16bf b = *(const v16bf*)(Bst + ((t << 5) + lane) * 16);
            acc[t] = wmma_bf16(a, b, acc[t]);
        }
        __syncthreads();
    }

#pragma unroll
    for (int t = 0; t < 4; ++t) {
        const int c = nbase + (t << 4) + r16;
        const float bv = bias[c];
#pragma unroll
        for (int r = 0; r < 8; ++r) {
            const int mo = mbase + ((lane >> 4) << 3) + r;
            if (mo < M_) dst[(size_t)mo * D_ + c] = acc[t][r] + bv;
        }
    }
}

// ---------------------------------------------------------------------------
extern "C" void kernel_launch(void* const* d_in, const int* in_sizes, int n_in,
                              void* d_out, int out_size, void* d_ws, size_t ws_size,
                              hipStream_t stream) {
    (void)in_sizes; (void)n_in; (void)out_size; (void)ws_size;

    const float* x     = (const float*)d_in[0];
    const float* l     = (const float*)d_in[1];
    /* d_in[2] = mask (unused by forward) */
    const float* Wqkv  = (const float*)d_in[3];
    const float* bqkv  = (const float*)d_in[4];
    const float* Wqkv1 = (const float*)d_in[5];
    const float* bqkv1 = (const float*)d_in[6];
    const float* c1w   = (const float*)d_in[7];
    const float* c1b   = (const float*)d_in[8];
    const float* bng   = (const float*)d_in[9];
    const float* bnb   = (const float*)d_in[10];
    const float* c2w   = (const float*)d_in[11];
    const float* c2b   = (const float*)d_in[12];
    const float* Wm    = (const float*)d_in[13];
    const float* bm    = (const float*)d_in[14];
    const float* Wm1   = (const float*)d_in[15];
    const float* bm1   = (const float*)d_in[16];

    // Workspace layout (bf16 elements): q[2], k[2], vT[2], headOut[2]
    bf16* wsQ  = (bf16*)d_ws;
    bf16* wsK  = wsQ  + 2 * SETQ;
    bf16* wsVt = wsK  + 2 * SETQ;
    bf16* wsO  = wsVt + 2 * SETQ;

    qkv_gemm_kernel<<<dim3(24, 129, 2), 128, 0, stream>>>(
        x, l, Wqkv, bqkv, Wqkv1, bqkv1, wsQ, wsK, wsVt);

    attn_kernel<<<dim3(17, B_ * H_), 128, 0, stream>>>(
        wsQ, wsK, wsVt, wsO, c1w, c1b, bng, bnb, c2w, c2b);

    mlp_gemm_kernel<<<dim3(8, 129, 2), 128, 0, stream>>>(
        wsO, Wm, bm, Wm1, bm1, (float*)d_out);
}
